// SamVisionAttention_58420145160729
// MI455X (gfx1250) — matrix operand order repaired and checked by reading, not compiled, not run on hardware
//
#include <hip/hip_runtime.h>

// ---------------------------------------------------------------------------
// SAM vision attention for MI455X (gfx1250, wave32, WMMA bf16 16x16x32).
// Pipeline: qkv GEMM -> relpos tables -> fused per-head attention -> proj GEMM.
// Heavy math via v_wmma_f32_16x16x32_bf16; raw bf16 tile staging via
// GLOBAL_LOAD_ASYNC_TO_LDS_B128 (ASYNCcnt) when the toolchain exposes it.
// ---------------------------------------------------------------------------

#define NUM_HEADS 16
#define HEAD_DIM  80
#define WIN       14
#define SEQ       196            // WIN*WIN
#define HIDDEN    1280
#define BATCH     256
#define BHTOT     (BATCH*NUM_HEADS)   // 4096
#define SPAD      208            // SEQ padded to 13*16
#define DPAD      96             // HEAD_DIM padded to 3*32
#define M1        (BATCH*SEQ)    // 50176 rows for both big GEMMs
#define SCALE_F   0.11180339887498949f   // 80^-0.5

typedef unsigned short u16;
typedef unsigned int   u32;
typedef __attribute__((ext_vector_type(16))) __bf16 v16bf;
typedef __attribute__((ext_vector_type(8)))  float  v8f;

union BF16Frag { v16bf v; u16 u[16]; uint4 q[2]; };
union F32Acc   { v8f v; float f[8]; };

__device__ __forceinline__ u16 f2bf(float f) {
    u32 u = __float_as_uint(f);
    u32 r = u + 0x7FFFu + ((u >> 16) & 1u);   // round-to-nearest-even
    return (u16)(r >> 16);
}
__device__ __forceinline__ float b2f(u16 u) {
    return __uint_as_float(((u32)u) << 16);
}
__device__ __forceinline__ u32 pack2(float a, float b) {
    return (u32)f2bf(a) | ((u32)f2bf(b) << 16);
}

// ---- CDNA5 async global->LDS copy (16B per lane), guarded ------------------
#if __has_builtin(__builtin_amdgcn_global_load_async_to_lds_b128)
#define HAVE_ASYNC_LDS 1
#else
#define HAVE_ASYNC_LDS 0
#endif

typedef int v4i128 __attribute__((vector_size(16)));   // matches builtin param

__device__ __forceinline__ void async_copy16(const void* g, void* l) {
#if HAVE_ASYNC_LDS
    __builtin_amdgcn_global_load_async_to_lds_b128(
        (__attribute__((address_space(1))) v4i128*)g,
        (__attribute__((address_space(3))) v4i128*)l, 0, 0);
#else
    *(uint4*)l = *(const uint4*)g;
#endif
}
__device__ __forceinline__ void async_wait0() {
#if HAVE_ASYNC_LDS
#if __has_builtin(__builtin_amdgcn_s_wait_asynccnt)
    __builtin_amdgcn_s_wait_asynccnt(0);
#else
    asm volatile("s_wait_asynccnt 0" ::: "memory");
#endif
#endif
}

// ---------------------------------------------------------------------------
// Kernel 1: qkv = hidden @ w_qkv^T + b_qkv  (M=50176, N=3840, K=1280)
// Output scattered to bf16 q/k/v workspaces [BH][SPAD][DPAD].
// Block tile 128x128, BK=32, 8 waves each computing 32x64.
// ---------------------------------------------------------------------------
#define BM  128
#define BN  128
#define BK  32
#define LDK 40   // padded LDS row pitch (bf16): 20-dword stride -> conflict free

__global__ __launch_bounds__(256)
void qkv_gemm_kernel(const float* __restrict__ X, const float* __restrict__ W,
                     const float* __restrict__ bias,
                     u16* __restrict__ qw, u16* __restrict__ kw, u16* __restrict__ vw)
{
    __shared__ u16 Alds[BM * LDK];
    __shared__ u16 Blds[BN * LDK];
    const int tid  = threadIdx.x;
    const int lane = tid & 31;
    const int wid  = tid >> 5;
    const int wm   = wid >> 1;   // 0..3  (32-row slice)
    const int wn   = wid & 1;    // 0..1  (64-col slice)
    const int m0   = blockIdx.x * BM;
    const int n0   = blockIdx.y * BN;

    F32Acc acc[2][4];
    for (int mt = 0; mt < 2; ++mt)
        for (int nt = 0; nt < 4; ++nt)
            for (int r = 0; r < 8; ++r) acc[mt][nt].f[r] = 0.f;

    const int kbA  = (lane < 16) ? 0 : 8;
    const int kbB  = (lane < 16) ? 0 : 8;
    const int frow = lane & 15;
    const int prow = tid >> 1, pcol = (tid & 1) << 4;   // prefetch pattern

    for (int k0 = 0; k0 < HIDDEN; k0 += BK) {
        if (k0 + BK < HIDDEN) {  // pull next K-panel toward GL2
            __builtin_prefetch(X + (size_t)(m0 + prow) * HIDDEN + k0 + BK + pcol, 0, 1);
            __builtin_prefetch(W + (size_t)(n0 + prow) * HIDDEN + k0 + BK + pcol, 0, 1);
        }
        // stage A (fp32 -> bf16) : 128x32
        for (int r = 0; r < 4; ++r) {
            int i = tid + r * 256;            // 0..1023 float4 chunks
            int row = i >> 3, c4 = (i & 7) << 2;
            float4 va = *(const float4*)(X + (size_t)(m0 + row) * HIDDEN + k0 + c4);
            *(u32*)&Alds[row * LDK + c4]     = pack2(va.x, va.y);
            *(u32*)&Alds[row * LDK + c4 + 2] = pack2(va.z, va.w);
        }
        // stage B (fp32 -> bf16) : 128 rows of w_qkv
        for (int r = 0; r < 4; ++r) {
            int i = tid + r * 256;
            int row = i >> 3, c4 = (i & 7) << 2;
            float4 vb = *(const float4*)(W + (size_t)(n0 + row) * HIDDEN + k0 + c4);
            *(u32*)&Blds[row * LDK + c4]     = pack2(vb.x, vb.y);
            *(u32*)&Blds[row * LDK + c4 + 2] = pack2(vb.z, vb.w);
        }
        __syncthreads();

        BF16Frag a[2], b[4];
        for (int mt = 0; mt < 2; ++mt) {
            int row = wm * 32 + mt * 16 + frow;
            a[mt].q[0] = *(const uint4*)&Alds[row * LDK + kbA];
            a[mt].q[1] = *(const uint4*)&Alds[row * LDK + kbA + 16];
        }
        for (int nt = 0; nt < 4; ++nt) {
            int row = wn * 64 + nt * 16 + frow;
            b[nt].q[0] = *(const uint4*)&Blds[row * LDK + kbB];
            b[nt].q[1] = *(const uint4*)&Blds[row * LDK + kbB + 16];
        }
        for (int mt = 0; mt < 2; ++mt)
            for (int nt = 0; nt < 4; ++nt)
                acc[mt][nt].v = __builtin_amdgcn_wmma_f32_16x16x32_bf16(
                    false, a[mt].v, false, b[nt].v, (short)0, acc[mt][nt].v, false, false);
        __syncthreads();
    }

    // epilogue: add bias, scatter bf16 into q/k/v [BH][SPAD][DPAD]
    // hoist index decompositions: (which,h,d) depends only on nt; (b,s) on (mt,r)
    const int mtop = (lane >> 4) << 3;
    const int ncol = lane & 15;
    u16*   dstp[4];
    size_t noff[4];
    float  bv[4];
    for (int nt = 0; nt < 4; ++nt) {
        int ng = n0 + wn * 64 + nt * 16 + ncol;
        int which = ng / HIDDEN, rem = ng - which * HIDDEN;
        int h = rem / HEAD_DIM, d = rem - h * HEAD_DIM;
        dstp[nt] = (which == 0) ? qw : (which == 1) ? kw : vw;
        noff[nt] = (size_t)h * (SPAD * DPAD) + d;
        bv[nt]   = bias[ng];
    }
    for (int mt = 0; mt < 2; ++mt)
        for (int r = 0; r < 8; ++r) {
            int mg = m0 + wm * 32 + mt * 16 + mtop + r;
            int bb = mg / SEQ, s = mg - bb * SEQ;
            size_t moff = (size_t)bb * (NUM_HEADS * SPAD * DPAD) + (size_t)s * DPAD;
            for (int nt = 0; nt < 4; ++nt)
                dstp[nt][moff + noff[nt]] = f2bf(acc[mt][nt].f[r] + bv[nt]);
        }
}

// ---------------------------------------------------------------------------
// Kernel 2: decomposed rel-pos tables
// ---------------------------------------------------------------------------
__global__ __launch_bounds__(256)
void relpos_kernel(const u16* __restrict__ q,
                   const float* __restrict__ Rh, const float* __restrict__ Rw,
                   float* __restrict__ relh, float* __restrict__ relw)
{
    const int bh = blockIdx.x;
    const u16* qb = q + (size_t)bh * SPAD * DPAD;
    for (int i = threadIdx.x; i < SEQ * 2 * WIN; i += 256) {
        int s  = i / (2 * WIN);
        int t  = i - s * (2 * WIN);
        int isW = (t >= WIN);
        int kk = isW ? (t - WIN) : t;
        int qh = s / WIN, qwc = s - qh * WIN;
        int ridx = (isW ? qwc : qh) - kk + (WIN - 1);
        const float* tab = (isW ? Rw : Rh) + ridx * HEAD_DIM;
        float sum = 0.f;
        for (int d = 0; d < HEAD_DIM; ++d)
            sum += b2f(qb[s * DPAD + d]) * tab[d];
        float* out = isW ? relw : relh;
        out[(size_t)bh * (SEQ * WIN) + s * WIN + kk] = sum;
    }
}

// ---------------------------------------------------------------------------
// Kernel 3: fused attention per head-batch. One block (8 waves) per bh.
// LDS: K [208][104] bf16, V^T [96][232] bf16, per-wave P scratch [16][232],
//      relh/relw tables. ~169 KB dynamic (WGP holds 320 KB).
// ---------------------------------------------------------------------------
#define NTILES 13
#define KROW   104
#define VROW   232
#define PROW   232

__global__ __launch_bounds__(256)
void attention_kernel(const u16* __restrict__ qg, const u16* __restrict__ kg,
                      const u16* __restrict__ vg,
                      const float* __restrict__ relh, const float* __restrict__ relw,
                      u16* __restrict__ out)
{
    extern __shared__ char smem[];
    u16*   kl = (u16*)smem;                        // [SPAD][KROW]
    u16*   vt = kl + SPAD * KROW;                  // [DPAD][VROW] (transposed V)
    u16*   pl = vt + DPAD * VROW;                  // 8 waves x [16][PROW]
    float* rh = (float*)(pl + 8 * 16 * PROW);
    float* rw = rh + SEQ * WIN;

    const int tid = threadIdx.x, lane = tid & 31, wid = tid >> 5;
    const int bh = blockIdx.x;
    const int bb = bh >> 4, hh = bh & 15;
    const size_t base = (size_t)bh * SPAD * DPAD;

    // stage K rows [j<196][d] via async 16B copies (no VGPR round trip)
    for (int i = tid; i < SEQ * 12; i += 256) {
        int j = i / 12, c = (i - j * 12) * 8;      // chunk start (u16 units)
        async_copy16(kg + base + j * DPAD + c, &kl[j * KROW + c]);
    }
    // zero K pad rows 196..207
    for (int i = tid; i < 12 * 48; i += 256) {
        int jj = i / 48, d = (i - jj * 48) * 2;
        *(u32*)&kl[(SEQ + jj) * KROW + d] = 0u;
    }
    // stage V transposed: vt[d][j] = v[j][d]
    for (int i = tid; i < SEQ * 48; i += 256) {
        int j = i / 48, d = (i - j * 48) * 2;
        u32 val = *(const u32*)(vg + base + j * DPAD + d);
        vt[d * VROW + j]       = (u16)(val & 0xFFFFu);
        vt[(d + 1) * VROW + j] = (u16)(val >> 16);
    }
    // zero V^T pad columns j = 196..231
    for (int i = tid; i < DPAD * 36; i += 256) {
        int d = i / 36, j = SEQ + (i - d * 36);
        vt[d * VROW + j] = 0;
    }
    for (int i = tid; i < SEQ * WIN; i += 256) {
        rh[i] = relh[(size_t)bh * (SEQ * WIN) + i];
        rw[i] = relw[(size_t)bh * (SEQ * WIN) + i];
    }
    async_wait0();
    __syncthreads();

    const int col  = lane & 15;
    const int kbA  = (lane < 16) ? 0 : 8;
    const int kbB  = (lane < 16) ? 0 : 8;
    const int mtop = (lane >> 4) << 3;
    u16* pw = pl + wid * 16 * PROW;

    for (int rt = wid; rt < NTILES; rt += 8) {
        // q A-fragments for the 3 K-steps (D padded to 96)
        BF16Frag qf[3];
        for (int ks = 0; ks < 3; ++ks) {
            const u16* src = qg + base + (size_t)(rt * 16 + col) * DPAD + ks * 32 + kbA;
            qf[ks].q[0] = *(const uint4*)(src);
            qf[ks].q[1] = *(const uint4*)(src + 16);
        }
        // scores: 16 rows x 208 cols
        F32Acc acc[NTILES];
        for (int nt = 0; nt < NTILES; ++nt)
            for (int r = 0; r < 8; ++r) acc[nt].f[r] = 0.f;
        for (int ks = 0; ks < 3; ++ks)
            for (int nt = 0; nt < NTILES; ++nt) {
                BF16Frag bf;
                const u16* src = &kl[(nt * 16 + col) * KROW + ks * 32 + kbB];
                bf.q[0] = *(const uint4*)src;
                bf.q[1] = *(const uint4*)(src + 16);
                acc[nt].v = __builtin_amdgcn_wmma_f32_16x16x32_bf16(
                    false, qf[ks].v, false, bf.v, (short)0, acc[nt].v, false, false);
            }
        // scale + decomposed rel-pos bias + column mask
        for (int nt = 0; nt < NTILES; ++nt) {
            int j = nt * 16 + col;
            int kh = j / WIN, kwc = j - kh * WIN;
            for (int r = 0; r < 8; ++r) {
                int sq = rt * 16 + mtop + r;
                float v = acc[nt].f[r];
                if (j < SEQ && sq < SEQ)
                    v = v * SCALE_F + rh[sq * WIN + kh] + rw[sq * WIN + kwc];
                else
                    v = -3.0e38f;
                acc[nt].f[r] = v;
            }
        }
        // softmax over 208 columns (reduce across the 16-lane column group)
        float rsum[8];
        for (int r = 0; r < 8; ++r) {
            float m = -3.4e38f;
            for (int nt = 0; nt < NTILES; ++nt) m = fmaxf(m, acc[nt].f[r]);
            for (int off = 1; off < 16; off <<= 1) m = fmaxf(m, __shfl_xor(m, off, 32));
            float s = 0.f;
            for (int nt = 0; nt < NTILES; ++nt) {
                float p = __expf(acc[nt].f[r] - m);
                acc[nt].f[r] = p;
                s += p;
            }
            for (int off = 1; off < 16; off <<= 1) s += __shfl_xor(s, off, 32);
            rsum[r] = s;
        }
        // write P (bf16) into per-wave scratch: C-layout -> A-layout
        for (int nt = 0; nt < NTILES; ++nt) {
            int j = nt * 16 + col;
            for (int r = 0; r < 8; ++r)
                pw[(mtop + r) * PROW + j] = f2bf(acc[nt].f[r]);
        }
        for (int e = lane; e < 256; e += 32) {  // zero pad cols 208..223
            int rr = e >> 4, cc = 208 + (e & 15);
            pw[rr * PROW + cc] = 0;
        }
        asm volatile("s_wait_dscnt 0" ::: "memory");  // wave-local LDS RAW

        BF16Frag pf[7];
        for (int js = 0; js < 7; ++js) {
            const u16* src = &pw[col * PROW + js * 32 + kbA];
            pf[js].q[0] = *(const uint4*)src;
            pf[js].q[1] = *(const uint4*)(src + 16);
        }
        // out(16x80) = P(16x224) @ V(224x80)
        for (int dt = 0; dt < 5; ++dt) {
            F32Acc o;
            for (int r = 0; r < 8; ++r) o.f[r] = 0.f;
            for (int js = 0; js < 7; ++js) {
                BF16Frag bf;
                const u16* src = &vt[(dt * 16 + col) * VROW + js * 32 + kbB];
                bf.q[0] = *(const uint4*)src;
                bf.q[1] = *(const uint4*)(src + 16);
                o.v = __builtin_amdgcn_wmma_f32_16x16x32_bf16(
                    false, pf[js].v, false, bf.v, (short)0, o.v, false, false);
            }
            for (int r = 0; r < 8; ++r) {
                int sq = rt * 16 + mtop + r;
                if (sq < SEQ) {
                    float val = o.f[r] / rsum[r];
                    out[((size_t)(bb * SEQ) + sq) * HIDDEN + hh * HEAD_DIM + dt * 16 + col]
                        = f2bf(val);
                }
            }
        }
    }
}

// ---------------------------------------------------------------------------
// Kernel 4: out = attn_out @ w_proj^T + b_proj  (M=50176, N=1280, K=1280)
// A already bf16 -> async B128 staged; B converted fp32->bf16. fp32 output.
// ---------------------------------------------------------------------------
__global__ __launch_bounds__(256)
void proj_gemm_kernel(const u16* __restrict__ A, const float* __restrict__ W,
                      const float* __restrict__ bias, float* __restrict__ out)
{
    __shared__ u16 Alds[BM * LDK];
    __shared__ u16 Blds[BN * LDK];
    const int tid  = threadIdx.x;
    const int lane = tid & 31;
    const int wid  = tid >> 5;
    const int wm   = wid >> 1;
    const int wn   = wid & 1;
    const int m0   = blockIdx.x * BM;
    const int n0   = blockIdx.y * BN;

    F32Acc acc[2][4];
    for (int mt = 0; mt < 2; ++mt)
        for (int nt = 0; nt < 4; ++nt)
            for (int r = 0; r < 8; ++r) acc[mt][nt].f[r] = 0.f;

    const int kbA  = (lane < 16) ? 0 : 8;
    const int kbB  = (lane < 16) ? 0 : 8;
    const int frow = lane & 15;
    const int prow = tid >> 1, pcol = (tid & 1) << 4;

    for (int k0 = 0; k0 < HIDDEN; k0 += BK) {
        if (k0 + BK < HIDDEN) {
            __builtin_prefetch(A + (size_t)(m0 + prow) * HIDDEN + k0 + BK + pcol, 0, 1);
            __builtin_prefetch(W + (size_t)(n0 + prow) * HIDDEN + k0 + BK + pcol, 0, 1);
        }
        for (int r = 0; r < 2; ++r) {           // A: async bf16 B128 copies
            int i = tid + r * 256;              // 0..511 chunks of 8 bf16
            int row = i >> 2, c8 = (i & 3) << 3;
            async_copy16(A + (size_t)(m0 + row) * HIDDEN + k0 + c8,
                         &Alds[row * LDK + c8]);
        }
        for (int r = 0; r < 4; ++r) {           // B: fp32 -> bf16
            int i = tid + r * 256;
            int row = i >> 3, c4 = (i & 7) << 2;
            float4 vb = *(const float4*)(W + (size_t)(n0 + row) * HIDDEN + k0 + c4);
            *(u32*)&Blds[row * LDK + c4]     = pack2(vb.x, vb.y);
            *(u32*)&Blds[row * LDK + c4 + 2] = pack2(vb.z, vb.w);
        }
        async_wait0();
        __syncthreads();

        BF16Frag a[2], b[4];
        for (int mt = 0; mt < 2; ++mt) {
            int row = wm * 32 + mt * 16 + frow;
            a[mt].q[0] = *(const uint4*)&Alds[row * LDK + kbA];
            a[mt].q[1] = *(const uint4*)&Alds[row * LDK + kbA + 16];
        }
        for (int nt = 0; nt < 4; ++nt) {
            int row = wn * 64 + nt * 16 + frow;
            b[nt].q[0] = *(const uint4*)&Blds[row * LDK + kbB];
            b[nt].q[1] = *(const uint4*)&Blds[row * LDK + kbB + 16];
        }
        for (int mt = 0; mt < 2; ++mt)
            for (int nt = 0; nt < 4; ++nt)
                acc[mt][nt].v = __builtin_amdgcn_wmma_f32_16x16x32_bf16(
                    false, a[mt].v, false, b[nt].v, (short)0, acc[mt][nt].v, false, false);
        __syncthreads();
    }

    const int mtop = (lane >> 4) << 3;
    const int ncol = lane & 15;
    for (int mt = 0; mt < 2; ++mt)
        for (int nt = 0; nt < 4; ++nt) {
            int ng = n0 + wn * 64 + nt * 16 + ncol;
            float bvv = bias[ng];
            for (int r = 0; r < 8; ++r) {
                int mg = m0 + wm * 32 + mt * 16 + mtop + r;
                out[(size_t)mg * HIDDEN + ng] = acc[mt][nt].f[r] + bvv;
            }
        }
}

// ---------------------------------------------------------------------------
extern "C" void kernel_launch(void* const* d_in, const int* in_sizes, int n_in,
                              void* d_out, int out_size, void* d_ws, size_t ws_size,
                              hipStream_t stream)
{
    const float* hidden = (const float*)d_in[0];
    const float* w_qkv  = (const float*)d_in[1];
    const float* b_qkv  = (const float*)d_in[2];
    const float* w_proj = (const float*)d_in[3];
    const float* b_proj = (const float*)d_in[4];
    const float* rel_h  = (const float*)d_in[5];
    const float* rel_w  = (const float*)d_in[6];

    char* ws = (char*)d_ws;
    const size_t QS = (size_t)BHTOT * SPAD * DPAD * sizeof(u16);   // per-tensor bf16
    const size_t RH = (size_t)BHTOT * SEQ * WIN * sizeof(float);
    u16*   qw   = (u16*)ws;
    u16*   kw   = (u16*)(ws + QS);
    u16*   vw   = (u16*)(ws + 2 * QS);
    float* relh = (float*)(ws + 3 * QS);
    float* relw = (float*)(ws + 3 * QS + RH);
    u16*   attn = (u16*)(ws + 3 * QS + 2 * RH);

    qkv_gemm_kernel<<<dim3(M1 / BM, 3 * HIDDEN / BN), 256, 0, stream>>>(
        hidden, w_qkv, b_qkv, qw, kw, vw);

    relpos_kernel<<<dim3(BHTOT), 256, 0, stream>>>(qw, rel_h, rel_w, relh, relw);

    const size_t smem2 =
        (size_t)(SPAD * KROW + DPAD * VROW + 8 * 16 * PROW) * sizeof(u16)
        + 2 * (size_t)SEQ * WIN * sizeof(float);   // ~169 KB
    attention_kernel<<<dim3(BHTOT), 256, smem2, stream>>>(
        qw, kw, vw, relh, relw, attn);

    proj_gemm_kernel<<<dim3(M1 / BM, HIDDEN / BN), 256, 0, stream>>>(
        attn, w_proj, b_proj, (float*)d_out);
}
